// SCNN_81707457839215
// MI455X (gfx1250) — matrix operand order repaired and checked by reading, not compiled, not run on hardware
//
#include <hip/hip_runtime.h>
#include <hip/hip_bf16.h>

typedef __attribute__((ext_vector_type(16))) _Float16 v16h;
typedef __attribute__((ext_vector_type(8)))  _Float16 v8h;
typedef __attribute__((ext_vector_type(8)))  float    v8f;
typedef __attribute__((ext_vector_type(2)))  float    v2f;

#define THRESH 0.5f
#define DECAY  0.2f
#define DELTA  0.5f
#define TSTEPS 20
#define BATCH  100
#define BPAD   128   // batch padded to 4x K=32 WMMA steps

// ---------------- workspace layout (float offsets) ----------------
// persistent state + zero row (zeroed once per launch):
#define OFF_C1M   0            // 1,176,000
#define OFF_C1S   1176000      // 1,176,000
#define OFF_C2M   2352000      // 784,000
#define OFF_C2S   3136000      // 784,000
#define OFF_H1M   3920000      // 30,000
#define OFF_H1S   3950000      // 30,000
#define OFF_H2M   3980000      // 1,000
#define OFF_H2S   3981000      // 1,000
#define OFF_H2SUM 3982000      // 1,000
#define OFF_CXTR  3983000      // 78,400
#define OFF_C1TR  4061400      // 1,176,000
#define OFF_C2TR  5237400      // 784,000
#define OFF_XTR   6021400      // 196,000
#define OFF_H1TR  6217400      // 30,000
#define OFF_ZERO  6247400      // 1,984 (dummy zero row for fc1 padding lanes)
#define STATE_F32 6249384
// fully-rewritten scratch (no init needed):
#define OFF_CX    6249384      // 78,400
#define OFF_X1    6327784      // 294,000
#define OFF_X2    6621784      // 196,000
#define F32_TOTAL 6817784
// f16 spike records, batch-innermost, padded (T, dim_pad, 128); zeroed per launch
#define HOFF_C1SUMM 0          // 20*16*128    = 40,960   (pres of wwc2, P=15->16)
#define HOFF_C2SUMM 40960      // 20*48*128    = 122,880  (posts of wwc2, Q=40->48)
#define HOFF_XSUMM  163840     // 20*1968*128  = 5,038,080 (pres of wwh1, P=1960->1968)
#define HOFF_H1SUMM 5201920    // 20*304*128   = 778,240  (posts of wwh1, Q=300->304)
#define H16_TOTAL   5980160

// output regions (cumulative float offsets in d_out)
#define OUT_XMEAN  1000
#define OUT_H1MEAN 197000
#define OUT_C1SUM  227000
#define OUT_C2SUM  1403000
#define OUT_CXSUM  2187000
#define OUT_WWC2   2265400
#define OUT_WWH1   2266000

__device__ __forceinline__ float urand(unsigned s) {
    s ^= s >> 17; s *= 0xed5ad4bbu;
    s ^= s >> 11; s *= 0xac4c1b51u;
    s ^= s >> 15; s *= 0x31848babu;
    s ^= s >> 14;
    return (float)(s >> 8) * (1.0f / 16777216.0f);
}

// -------- Bernoulli encoding + input trace --------
__global__ void cx_kernel(const float* __restrict__ input, float* cx, float* cxtr, unsigned t) {
    int i = blockIdx.x * 256 + threadIdx.x;
    if (i >= BATCH * 784) return;
    float u = urand(i * 2654435761u + t * 0x9e3779b9u + 12345u);
    float s = (input[i] > u) ? 1.0f : 0.0f;
    cx[i] = s;
    cxtr[i] = DELTA * cxtr[i] + s;
}

// -------- conv1 (1->15, 3x3, pad 1) + LIF + trace + channel-sum record --------
__global__ void conv1_kernel(const float* __restrict__ cx, const float* __restrict__ w1,
                             const float* __restrict__ b1, float* c1m, float* c1s,
                             float* c1tr, _Float16* c1recT) {
    int bc = blockIdx.x;            // b*15 + c
    int b = bc / 15, c = bc % 15;
    float w[9];
#pragma unroll
    for (int i = 0; i < 9; ++i) w[i] = w1[c * 9 + i];
    float bias = b1[c];
    const float* cxb = cx + b * 784;
    float lsum = 0.0f;
    for (int pix = threadIdx.x; pix < 784; pix += 256) {
        int oh = pix / 28, ow = pix % 28;
        float con = bias;
#pragma unroll
        for (int kh = 0; kh < 3; ++kh) {
            int ih = oh + kh - 1;
            if (ih < 0 || ih >= 28) continue;
#pragma unroll
            for (int kw = 0; kw < 3; ++kw) {
                int iw = ow + kw - 1;
                if (iw < 0 || iw >= 28) continue;
                con += cxb[ih * 28 + iw] * w[kh * 3 + kw];
            }
        }
        int gi = bc * 784 + pix;
        float m = c1m[gi] * DECAY * (1.0f - c1s[gi]) + con;
        float s = (m > THRESH) ? 1.0f : 0.0f;
        c1m[gi] = m; c1s[gi] = s;
        c1tr[gi] = DELTA * c1tr[gi] + s;
        lsum += s;
    }
    __shared__ float red[256];
    red[threadIdx.x] = lsum;
    __syncthreads();
    for (int off = 128; off > 0; off >>= 1) {
        if (threadIdx.x < off) red[threadIdx.x] += red[threadIdx.x + off];
        __syncthreads();
    }
    if (threadIdx.x == 0) c1recT[c * BPAD + b] = (_Float16)red[0];
}

// -------- 2x2 mean pool of c1s -> x1 (B,15,14,14) --------
__global__ void pool1_kernel(const float* __restrict__ c1s, float* x1) {
    int i = blockIdx.x * 256 + threadIdx.x;
    if (i >= BATCH * 15 * 196) return;
    int b = i / 2940, r = i % 2940, c = r / 196, q = r % 196;
    int oh = q / 14, ow = q % 14;
    const float* src = c1s + ((b * 15 + c) * 28 + oh * 2) * 28 + ow * 2;
    x1[i] = 0.25f * (src[0] + src[1] + src[28] + src[29]);
}

// -------- conv2 (15->40, 3x3, pad 1) via LDS tile + LIF + record --------
__global__ void conv2_kernel(const float* __restrict__ x1, const float* __restrict__ w2,
                             const float* __restrict__ b2, float* c2m, float* c2s,
                             float* c2tr, _Float16* c2recT) {
    int bc = blockIdx.x;            // b*40 + c
    int b = bc / 40, c = bc % 40;
    __shared__ float tile[15 * 256];   // 15 chans, 16x16 padded
    __shared__ float wsh[136];
    __shared__ float red[256];
    int tid = threadIdx.x;
    const float* x1b = x1 + b * 2940;
    for (int i = tid; i < 15 * 256; i += 256) {
        int ic = i >> 8, rem = i & 255;
        int hh = (rem >> 4) - 1, ww = (rem & 15) - 1;
        float v = 0.0f;
        if (hh >= 0 && hh < 14 && ww >= 0 && ww < 14) v = x1b[ic * 196 + hh * 14 + ww];
        tile[i] = v;
    }
    for (int i = tid; i < 135; i += 256) wsh[i] = w2[c * 135 + i];
    __syncthreads();
    float lsum = 0.0f;
    if (tid < 196) {
        int oh = tid / 14, ow = tid % 14;
        float con = b2[c];
#pragma unroll
        for (int ic = 0; ic < 15; ++ic)
#pragma unroll
            for (int kh = 0; kh < 3; ++kh)
#pragma unroll
                for (int kw = 0; kw < 3; ++kw)
                    con += tile[ic * 256 + (oh + kh) * 16 + (ow + kw)] * wsh[ic * 9 + kh * 3 + kw];
        int gi = bc * 196 + tid;
        float m = c2m[gi] * DECAY * (1.0f - c2s[gi]) + con;
        float s = (m > THRESH) ? 1.0f : 0.0f;
        c2m[gi] = m; c2s[gi] = s;
        c2tr[gi] = DELTA * c2tr[gi] + s;
        lsum = s;
    }
    red[tid] = lsum;
    __syncthreads();
    for (int off = 128; off > 0; off >>= 1) {
        if (tid < off) red[tid] += red[tid + off];
        __syncthreads();
    }
    if (tid == 0) c2recT[c * BPAD + b] = (_Float16)red[0];
}

// -------- pool2 -> x2 (B,1960) + trace + transposed f16 record --------
__global__ void pool2_kernel(const float* __restrict__ c2s, float* x2, float* xtr,
                             _Float16* xrecT) {
    int i = blockIdx.x * 256 + threadIdx.x;
    if (i >= BATCH * 1960) return;
    int b = i / 1960, r = i % 1960, c = r / 49, q = r % 49;
    int oh = q / 7, ow = q % 7;
    const float* src = c2s + ((b * 40 + c) * 14 + oh * 2) * 14 + ow * 2;
    float v = 0.25f * (src[0] + src[1] + src[14] + src[15]);
    x2[i] = v;
    xtr[i] = DELTA * xtr[i] + v;
    xrecT[r * BPAD + b] = (_Float16)v;
}

// -------- fc1: (100x1960)@(1960x300)^T via v_wmma_f32_16x16x4_f32, fused LIF --------
// Padding lanes read a zeroed dummy row -> inner loop is branchless b64 loads + wmma.
__global__ void __launch_bounds__(32)
fc1_wmma_kernel(const float* __restrict__ x2, const float* __restrict__ wf1,
                const float* __restrict__ bf1, const float* __restrict__ zrow,
                float* h1m, float* h1s, float* h1tr, _Float16* h1recT) {
    int lane = threadIdx.x;
    int m0 = blockIdx.x * 16;   // batch tile (7 tiles for 100)
    int n0 = blockIdx.y * 16;   // neuron tile (19 tiles for 300)
    v8f acc;
#pragma unroll
    for (int i = 0; i < 8; ++i) acc[i] = 0.0f;
    int aM = m0 + (lane & 15);
    int bN = n0 + (lane & 15);
    const float* arow = (aM < BATCH) ? (x2 + aM * 1960) : zrow;
    const float* brow = (bN < 300)   ? (wf1 + bN * 1960) : zrow;
    int koff = (lane >> 4) << 1;
    arow += koff;
    brow += koff;
#pragma unroll 4
    for (int kk = 0; kk < 1960; kk += 4) {
        v2f a  = *(const v2f*)(arow + kk);
        v2f bv = *(const v2f*)(brow + kk);
        acc = __builtin_amdgcn_wmma_f32_16x16x4_f32(false, a, false, bv, (short)0, acc,
                                                    false, false);
    }
    int hi = lane >> 4;
#pragma unroll
    for (int r = 0; r < 8; ++r) {
        int M = m0 + r + 8 * hi;
        int N = n0 + (lane & 15);
        if (M < BATCH && N < 300) {
            float con = acc[r] + bf1[N];
            int gi = M * 300 + N;
            float m = h1m[gi] * DECAY * (1.0f - h1s[gi]) + con;
            float s = (m > THRESH) ? 1.0f : 0.0f;
            h1m[gi] = m; h1s[gi] = s;
            h1tr[gi] = DELTA * h1tr[gi] + s;
            h1recT[N * BPAD + M] = (_Float16)s;
        }
    }
}

// -------- fc2 (300->10) + LIF + sum --------
__global__ void fc2_kernel(const float* __restrict__ h1s, const float* __restrict__ wf2,
                           const float* __restrict__ bf2, float* h2m, float* h2s,
                           float* h2sum) {
    int i = blockIdx.x * 256 + threadIdx.x;
    if (i >= BATCH * 10) return;
    int b = i / 10, q = i % 10;
    float con = bf2[q];
    const float* hr = h1s + b * 300;
    const float* wr = wf2 + q * 300;
    for (int j = 0; j < 300; ++j) con += hr[j] * wr[j];
    float m = h2m[i] * DECAY * (1.0f - h2s[i]) + con;
    float s = (m > THRESH) ? 1.0f : 0.0f;
    h2m[i] = m; h2s[i] = s;
    h2sum[i] += s;
}

// -------- STDP tile matmul: fully branchless, contiguous fragment loads --------
// postsT: (Qpad,128) f16 row-major batch-innermost; presT: (Ppad,128).
// A[m,k]=postsT[m*128+k] -> per lane two contiguous 8-half (16B) chunks.
// B[k,n]=presT[n*128+k]  -> per lane one contiguous 16-half (32B) chunk.
__device__ __forceinline__ v8f stdp_mm(const _Float16* __restrict__ postsT,
                                       const _Float16* __restrict__ presT,
                                       int q0, int p0, int lane) {
    v8f acc;
#pragma unroll
    for (int i = 0; i < 8; ++i) acc[i] = 0.0f;
    int hi = lane >> 4;
    const _Float16* arow = postsT + (q0 + (lane & 15)) * BPAD + hi * 16;
    const _Float16* brow = presT + (p0 + (lane & 15)) * BPAD + hi * 16;
#pragma unroll
    for (int ks = 0; ks < BPAD; ks += 32) {
        v8h alo = *(const v8h*)(arow + ks);
        v8h ahi = *(const v8h*)(arow + ks + 8);
        v16h a;
#pragma unroll
        for (int h = 0; h < 8; ++h) { a[h] = alo[h]; a[h + 8] = ahi[h]; }
        v16h bv = *(const v16h*)(brow + ks);
        acc = __builtin_amdgcn_wmma_f32_16x16x32_f16(false, a, false, bv, (short)0, acc,
                                                     false, false);
    }
    return acc;
}

__global__ void __launch_bounds__(32)
stdp_kernel(const _Float16* __restrict__ pres16, const _Float16* __restrict__ posts16,
            float* __restrict__ out, int Q, int P, int qstride, int pstride) {
    int lane = threadIdx.x;
    int p0 = blockIdx.x * 16;
    int q0 = blockIdx.y * 16;
    float e[8];
#pragma unroll
    for (int i = 0; i < 8; ++i) e[i] = 0.0f;
    for (int n = 0; n < 14; ++n) {
        int tq = 6 + n;
        for (int k = 0; k < 5; ++k) {
            int tp = tq - (k + 1);
            int idx = n * 10 + k;
            float ew = __expf(-(float)(k + 1) * 0.05f);
            float cpos = 0.925f * ew * __powf(0.8f, (float)(139 - idx));
            float cneg = 0.100f * ew * __powf(0.8f, (float)(139 - idx - 5));
            v8f ap = stdp_mm(posts16 + tq * qstride, pres16 + tp * pstride, q0, p0, lane);
            v8f an = stdp_mm(posts16 + tp * qstride, pres16 + tq * pstride, q0, p0, lane);
#pragma unroll
            for (int r = 0; r < 8; ++r) {
                e[r] += (ap[r] > 50.0f) ? cpos : 0.0f;
                e[r] -= (an[r] > 50.0f) ? cneg : 0.0f;
            }
        }
    }
    int hi = lane >> 4;
#pragma unroll
    for (int r = 0; r < 8; ++r) {
        int qq = q0 + r + 8 * hi;
        int pp = p0 + (lane & 15);
        if (qq < Q && pp < P) out[qq * P + pp] = e[r];
    }
}

// -------- finalize: outputs, xmean, h1mean, c1sum, c2sum, cxsum --------
__global__ void finalize_kernel(const float* __restrict__ h2sum, const float* __restrict__ xtr,
                                const float* __restrict__ h1tr, const float* __restrict__ c1tr,
                                const float* __restrict__ c2tr, const float* __restrict__ cxtr,
                                float* __restrict__ out) {
    int i = blockIdx.x * 256 + threadIdx.x;
    const float inv = 1.0f / (float)TSTEPS;
    if (i < 1000)          out[i] = h2sum[i] * inv;
    else if (i < 197000)   out[i] = xtr[i - OUT_XMEAN] * inv;
    else if (i < 227000)   out[i] = h1tr[i - OUT_H1MEAN] * inv;
    else if (i < 1403000)  out[i] = c1tr[i - OUT_C1SUM] * inv;
    else if (i < 2187000)  out[i] = c2tr[i - OUT_C2SUM] * inv;
    else if (i < 2265400)  out[i] = cxtr[i - OUT_CXSUM] * inv;
}

extern "C" void kernel_launch(void* const* d_in, const int* in_sizes, int n_in,
                              void* d_out, int out_size, void* d_ws, size_t ws_size,
                              hipStream_t stream) {
    const float* input = (const float*)d_in[0];
    const float* w1  = (const float*)d_in[4];
    const float* b1  = (const float*)d_in[5];
    const float* w2  = (const float*)d_in[6];
    const float* b2  = (const float*)d_in[7];
    const float* wf1 = (const float*)d_in[8];
    const float* bf1 = (const float*)d_in[9];
    const float* wf2 = (const float*)d_in[10];
    const float* bf2 = (const float*)d_in[11];

    float* ws = (float*)d_ws;
    _Float16* h16 = (_Float16*)(ws + F32_TOTAL);
    float* out = (float*)d_out;

    // zero persistent state + zero-row each call; zero f16 records so batch/dim
    // padding stays zero (graph-capture safe)
    hipMemsetAsync(d_ws, 0, (size_t)STATE_F32 * sizeof(float), stream);
    hipMemsetAsync((void*)h16, 0, (size_t)H16_TOTAL * sizeof(_Float16), stream);

    for (unsigned t = 0; t < TSTEPS; ++t) {
        cx_kernel<<<(BATCH * 784 + 255) / 256, 256, 0, stream>>>(
            input, ws + OFF_CX, ws + OFF_CXTR, t);
        conv1_kernel<<<BATCH * 15, 256, 0, stream>>>(
            ws + OFF_CX, w1, b1, ws + OFF_C1M, ws + OFF_C1S, ws + OFF_C1TR,
            h16 + HOFF_C1SUMM + t * 16 * BPAD);
        pool1_kernel<<<(BATCH * 15 * 196 + 255) / 256, 256, 0, stream>>>(
            ws + OFF_C1S, ws + OFF_X1);
        conv2_kernel<<<BATCH * 40, 256, 0, stream>>>(
            ws + OFF_X1, w2, b2, ws + OFF_C2M, ws + OFF_C2S, ws + OFF_C2TR,
            h16 + HOFF_C2SUMM + t * 48 * BPAD);
        pool2_kernel<<<(BATCH * 1960 + 255) / 256, 256, 0, stream>>>(
            ws + OFF_C2S, ws + OFF_X2, ws + OFF_XTR,
            h16 + HOFF_XSUMM + t * 1968 * BPAD);
        fc1_wmma_kernel<<<dim3(7, 19), 32, 0, stream>>>(
            ws + OFF_X2, wf1, bf1, ws + OFF_ZERO, ws + OFF_H1M, ws + OFF_H1S,
            ws + OFF_H1TR, h16 + HOFF_H1SUMM + t * 304 * BPAD);
        fc2_kernel<<<(BATCH * 10 + 255) / 256, 256, 0, stream>>>(
            ws + OFF_H1S, wf2, bf2, ws + OFF_H2M, ws + OFF_H2S, ws + OFF_H2SUM);
    }

    finalize_kernel<<<(2265400 + 255) / 256, 256, 0, stream>>>(
        ws + OFF_H2SUM, ws + OFF_XTR, ws + OFF_H1TR, ws + OFF_C1TR, ws + OFF_C2TR,
        ws + OFF_CXTR, out);

    // wwc2: Q=40 (3 tiles), P=15 (1 tile)
    stdp_kernel<<<dim3(1, 3), 32, 0, stream>>>(
        h16 + HOFF_C1SUMM, h16 + HOFF_C2SUMM, out + OUT_WWC2, 40, 15,
        48 * BPAD, 16 * BPAD);
    // wwh1: Q=300 (19 tiles), P=1960 (123 tiles)
    stdp_kernel<<<dim3(123, 19), 32, 0, stream>>>(
        h16 + HOFF_XSUMM, h16 + HOFF_H1SUMM, out + OUT_WWH1, 300, 1960,
        304 * BPAD, 1968 * BPAD);
}
